// GraphFeatureExtractor_14035953123570
// MI455X (gfx1250) — compile-verified, matrix operand order checked by reading, not run on hardware
//
#include <hip/hip_runtime.h>

// ---------------------------------------------------------------------------
// GAT feature extractor for MI455X (gfx1250, wave32).
// Dense node transforms: v_wmma_f32_16x16x32_bf16, one wave = 16x64 output
// tile (4 accumulators share one A fragment -> 2 A + 8 B b128 loads per
// 4 WMMAs). Edge softmax/aggregation: L2-resident gathers + f32/u32 atomics
// (h matrices are 20-40MB -> fit in the 192MB L2).
// ---------------------------------------------------------------------------

typedef __attribute__((ext_vector_type(16))) __bf16 v16bf;
typedef __attribute__((ext_vector_type(8)))  float  v8f;

#define WG 256

static inline unsigned gblocks(long long n) { return (unsigned)((n + WG - 1) / WG); }

// ------------------------------- utility -----------------------------------

__global__ void fill_f32(float* __restrict__ p, float v, long long n) {
  long long i = (long long)blockIdx.x * blockDim.x + threadIdx.x;
  if (i < n) p[i] = v;
}

__global__ void fill_u32(unsigned* __restrict__ p, unsigned v, long long n) {
  long long i = (long long)blockIdx.x * blockDim.x + threadIdx.x;
  if (i < n) p[i] = v;
}

// f32 -> bf16 feature copy with zero padding of K to Kpad (layer 0 only)
__global__ void cvt_feats(const float* __restrict__ in, __bf16* __restrict__ out,
                          int N, int K, int Kpad) {
  long long i = (long long)blockIdx.x * blockDim.x + threadIdx.x;
  long long total = (long long)N * Kpad;
  if (i >= total) return;
  int k = (int)(i % Kpad);
  long long r = i / Kpad;
  float v = (k < K) ? in[r * K + k] : 0.0f;
  out[i] = (__bf16)v;
}

// W[K, Nout] f32 -> Wt[Nout, Kpad] bf16 (transposed, zero padded)
__global__ void cvt_weights(const float* __restrict__ W, __bf16* __restrict__ Wt,
                            int K, int Kpad, int Nout) {
  long long i = (long long)blockIdx.x * blockDim.x + threadIdx.x;
  long long total = (long long)Nout * Kpad;
  if (i >= total) return;
  int k = (int)(i % Kpad);
  int n = (int)(i / Kpad);
  float v = (k < K) ? W[(long long)k * Nout + n] : 0.0f;
  Wt[i] = (__bf16)v;
}

// ------------------------------ WMMA GEMM ----------------------------------
// C[M,Nout] = A[M,K] * Bt[Nout,K]^T   (A,Bt bf16 row-major, C f32)
// One wave computes a 16x64 tile: 4 accumulators share one 16x32 A fragment
// per K-step of 32. Nout must be a multiple of 64; M a multiple of 16.
// A 16x32 bf16 layout: lanes 0-15 -> M=lane, K in {0..7,16..23};
//                      lanes 16-31 -> M=lane-16, K in {8..15,24..31}.
// B mirrored with N in place of M (Bt pre-transposed so loads are b128).
__global__ __launch_bounds__(WG)
void gemm_bf16_wmma(const __bf16* __restrict__ A, const __bf16* __restrict__ Bt,
                    float* __restrict__ C, int M, int K, int Nout) {
  int wave = (int)(((long long)blockIdx.x * blockDim.x + threadIdx.x) >> 5);
  int lane = threadIdx.x & 31;
  int tilesN4 = Nout >> 6;                     // N-tiles of 64
  int tM = wave / tilesN4;
  int tN = wave - tM * tilesN4;
  if (tM * 16 >= M) return;                    // wave-uniform: EXEC stays all-1s

  int r    = lane & 15;
  int koff = (lane >> 4) << 3;                 // 0 or 8
  const __bf16* Ap = A  + (long long)(tM * 16 + r) * K;
  const __bf16* Bp = Bt + (long long)(tN * 64 + r) * K;

  union U { v16bf v; uint4 q[2]; };
  v8f acc[4] = {};
  for (int k0 = 0; k0 < K; k0 += 32) {
    U a;
    a.q[0] = *(const uint4*)(Ap + k0 + koff);
    a.q[1] = *(const uint4*)(Ap + k0 + 16 + koff);
#pragma unroll
    for (int j = 0; j < 4; ++j) {
      U b;
      const __bf16* bp = Bp + (long long)j * 16 * K + k0;
      b.q[0] = *(const uint4*)(bp + koff);
      b.q[1] = *(const uint4*)(bp + 16 + koff);
      acc[j] = __builtin_amdgcn_wmma_f32_16x16x32_bf16(
          /*neg_a=*/false, a.v, /*neg_b=*/false, b.v,
          /*c_mod=*/(short)0, acc[j], /*reuse_a=*/false, /*reuse_b=*/false);
    }
  }

  // C/D layout: VGPR i -> M = i (lanes 0-15) or i+8 (lanes 16-31); N = lane&15
  int mb = tM * 16 + ((lane >> 4) << 3);
#pragma unroll
  for (int j = 0; j < 4; ++j) {
    int n = tN * 64 + j * 16 + (lane & 15);
#pragma unroll
    for (int i = 0; i < 8; ++i)
      C[(long long)(mb + i) * Nout + n] = acc[j][i];
  }
}

// --------------------------- attention kernels -----------------------------

// es[n,h] = sum_c hlin[n, h*C+c]*as[h,c] ;  ed likewise
__global__ void node_attn(const float* __restrict__ hlin,
                          const float* __restrict__ as_, const float* __restrict__ ad_,
                          float* __restrict__ es, float* __restrict__ ed,
                          int N, int C, int HC) {
  long long t = (long long)blockIdx.x * blockDim.x + threadIdx.x;
  if (t >= (long long)N * 4) return;
  int h = (int)(t & 3);
  long long n = t >> 2;
  const float* hp = hlin + n * HC + h * C;
  const float* ap = as_ + h * C;
  const float* dp = ad_ + h * C;
  float s = 0.f, d = 0.f;
  for (int c = 0; c < C; ++c) { float v = hp[c]; s += v * ap[c]; d += v * dp[c]; }
  es[t] = s; ed[t] = d;
}

__device__ __forceinline__ unsigned f2ord(float f) {
  unsigned u = __float_as_uint(f);
  return (u & 0x80000000u) ? ~u : (u | 0x80000000u);
}
__device__ __forceinline__ float ord2f(unsigned u) {
  return __uint_as_float((u & 0x80000000u) ? (u & 0x7FFFFFFFu) : ~u);
}

__device__ __forceinline__ void edge_endpoints(const int* __restrict__ ei,
                                               long long e, int E, int* src, int* dst) {
  if (e < E) { *src = ei[e]; *dst = ei[E + e]; }          // edge_index[0]/[1]
  else       { *src = *dst = (int)(e - E); }               // self-loop
}

// pass 1: segment max of leakyrelu(e) over dst, ordered-uint atomicMax
__global__ void edge_max(const int* __restrict__ ei, int E, long long EE,
                         const float* __restrict__ es, const float* __restrict__ ed,
                         unsigned* __restrict__ emax) {
  long long t = (long long)blockIdx.x * blockDim.x + threadIdx.x;
  if (t >= EE * 4) return;
  int h = (int)(t & 3);
  long long e = t >> 2;
  int src, dst; edge_endpoints(ei, e, E, &src, &dst);
  float v = es[(long long)src * 4 + h] + ed[(long long)dst * 4 + h];
  v = v > 0.f ? v : 0.2f * v;
  atomicMax(&emax[(long long)dst * 4 + h], f2ord(v));
}

// pass 2: ex = exp(e - max), store per-edge, accumulate denom
__global__ void edge_exp(const int* __restrict__ ei, int E, long long EE,
                         const float* __restrict__ es, const float* __restrict__ ed,
                         const unsigned* __restrict__ emax,
                         float* __restrict__ denom, float* __restrict__ exf) {
  long long t = (long long)blockIdx.x * blockDim.x + threadIdx.x;
  if (t >= EE * 4) return;
  int h = (int)(t & 3);
  long long e = t >> 2;
  int src, dst; edge_endpoints(ei, e, E, &src, &dst);
  float v = es[(long long)src * 4 + h] + ed[(long long)dst * 4 + h];
  v = v > 0.f ? v : 0.2f * v;
  float m = ord2f(emax[(long long)dst * 4 + h]);
  float ex = expf(v - m);
  exf[t] = ex;
  atomicAdd(&denom[(long long)dst * 4 + h], ex);
}

// pass 2b: normalize once per (edge, head) so the fat aggregate pass
// does not re-gather denom and divide per 4-channel thread
__global__ void edge_alpha(const int* __restrict__ ei, int E, long long EE,
                           const float* __restrict__ denom, float* __restrict__ exf) {
  long long t = (long long)blockIdx.x * blockDim.x + threadIdx.x;
  if (t >= EE * 4) return;
  int h = (int)(t & 3);
  long long e = t >> 2;
  int src, dst; edge_endpoints(ei, e, E, &src, &dst);
  exf[t] = exf[t] / (denom[(long long)dst * 4 + h] + 1e-16f);
}

// pass 3: out[dst] += alpha * hlin[src]; one thread = 4 channels (float4 gather)
__global__ void edge_agg(const int* __restrict__ ei, int E, long long EE, int C, int HC,
                         const float* __restrict__ alpha_,
                         const float* __restrict__ hlin, float* __restrict__ out) {
  long long t = (long long)blockIdx.x * blockDim.x + threadIdx.x;
  int cg = C >> 2;
  long long total = EE * 4 * cg;
  if (t >= total) return;
  int c4 = (int)(t % cg);
  long long tmp = t / cg;
  int h = (int)(tmp & 3);
  long long e = tmp >> 2;
  int src, dst; edge_endpoints(ei, e, E, &src, &dst);
  float alpha = alpha_[e * 4 + h];
  int base = h * C + c4 * 4;
  const float4 v = *(const float4*)(hlin + (long long)src * HC + base);
  float* o = out + (long long)dst * HC + base;
  atomicAdd(o + 0, alpha * v.x);
  atomicAdd(o + 1, alpha * v.y);
  atomicAdd(o + 2, alpha * v.z);
  atomicAdd(o + 3, alpha * v.w);
}

// bias + (optional) relu, and emit bf16 copy for the next layer's GEMM
__global__ void bias_act(float* __restrict__ h, const float* __restrict__ b,
                         __bf16* __restrict__ hbf,
                         long long total, int HC, int doRelu) {
  long long t = (long long)blockIdx.x * blockDim.x + threadIdx.x;
  if (t >= total) return;
  int c = (int)(t % HC);
  float v = h[t] + b[c];
  if (doRelu) v = fmaxf(v, 0.f);
  h[t] = v;
  if (hbf) hbf[t] = (__bf16)v;
}

// ------------------------------- pooling / head ----------------------------

__global__ void pool_cnt(const int* __restrict__ batch, float* __restrict__ cnt, int N) {
  int t = blockIdx.x * blockDim.x + threadIdx.x;
  if (t < N) atomicAdd(&cnt[batch[t]], 1.0f);
}

__global__ void pool_add(const float* __restrict__ h, const int* __restrict__ batch,
                         float* __restrict__ pooled, int N) {
  long long t = (long long)blockIdx.x * blockDim.x + threadIdx.x;
  if (t >= (long long)N * 128) return;               // 512 / 4
  int c4 = (int)(t % 128);
  int n = (int)(t / 128);
  int g = batch[n];
  const float4 v = *(const float4*)(h + (long long)n * 512 + c4 * 4);
  float* p = pooled + (long long)g * 512 + c4 * 4;
  atomicAdd(p + 0, v.x); atomicAdd(p + 1, v.y);
  atomicAdd(p + 2, v.z); atomicAdd(p + 3, v.w);
}

// LayerNorm over D=512 per row; optional mean-pool divide by cnt[g]. 256 thr/row.
__global__ __launch_bounds__(256)
void ln_row(const float* __restrict__ in, const float* __restrict__ cnt,
            const float* __restrict__ gam, const float* __restrict__ bet,
            float* __restrict__ out, int D) {
  __shared__ float red[256];
  int g = blockIdx.x, t = threadIdx.x;
  float scale = 1.0f;
  if (cnt) scale = 1.0f / fmaxf(cnt[g], 1.0f);
  float v0 = in[(long long)g * D + t] * scale;
  float v1 = in[(long long)g * D + t + 256] * scale;
  red[t] = v0 + v1; __syncthreads();
  for (int o = 128; o > 0; o >>= 1) { if (t < o) red[t] += red[t + o]; __syncthreads(); }
  float mean = red[0] / (float)D;
  __syncthreads();
  float d0 = v0 - mean, d1 = v1 - mean;
  red[t] = d0 * d0 + d1 * d1; __syncthreads();
  for (int o = 128; o > 0; o >>= 1) { if (t < o) red[t] += red[t + o]; __syncthreads(); }
  float rs = rsqrtf(red[0] / (float)D + 1e-5f);
  out[(long long)g * D + t]       = d0 * rs * gam[t] + bet[t];
  out[(long long)g * D + t + 256] = d1 * rs * gam[t + 256] + bet[t + 256];
}

// z2 = relu(z1 @ fcW + fcb) ; tiny (64x512x512)
__global__ void fc_relu(const float* __restrict__ z, const float* __restrict__ W,
                        const float* __restrict__ b, float* __restrict__ out, int D) {
  int t = blockIdx.x * blockDim.x + threadIdx.x;
  if (t >= 64 * D) return;
  int g = t / D, o = t % D;
  float s = b[o];
  const float* zr = z + (long long)g * D;
  for (int i = 0; i < D; ++i) s += zr[i] * W[(long long)i * D + o];
  out[t] = fmaxf(s, 0.f);
}

// ------------------------------- host side ---------------------------------

extern "C" void kernel_launch(void* const* d_in, const int* in_sizes, int n_in,
                              void* d_out, int out_size, void* d_ws, size_t ws_size,
                              hipStream_t stream) {
  const float* x     = (const float*)d_in[0];
  const int*   ei    = (const int*)d_in[1];
  const int*   batch = (const int*)d_in[2];
  const float* W_[4]  = {(const float*)d_in[3],  (const float*)d_in[7],
                         (const float*)d_in[11], (const float*)d_in[15]};
  const float* As[4]  = {(const float*)d_in[4],  (const float*)d_in[8],
                         (const float*)d_in[12], (const float*)d_in[16]};
  const float* Ad[4]  = {(const float*)d_in[5],  (const float*)d_in[9],
                         (const float*)d_in[13], (const float*)d_in[17]};
  const float* Bs[4]  = {(const float*)d_in[6],  (const float*)d_in[10],
                         (const float*)d_in[14], (const float*)d_in[18]};
  const float* ln1g = (const float*)d_in[19];
  const float* ln1b = (const float*)d_in[20];
  const float* fcW  = (const float*)d_in[21];
  const float* fcb  = (const float*)d_in[22];
  const float* ln2g = (const float*)d_in[23];
  const float* ln2b = (const float*)d_in[24];

  const int N  = in_sizes[2];          // 20000
  const int E  = in_sizes[1] / 2;      // 320000
  const long long EE = (long long)E + N;
  const int G = 64, OUT = 512;

  // workspace carve-out (256B aligned)
  size_t off = 0;
  auto take = [&](size_t bytes) -> void* {
    void* p = (char*)d_ws + off;
    off += (bytes + 255) & ~(size_t)255;
    return p;
  };
  float*    hA     = (float*)   take((size_t)N * 512 * 4);   // agg out / layer input
  float*    hB     = (float*)   take((size_t)N * 512 * 4);   // GEMM output
  __bf16*   hbf    = (__bf16*)  take((size_t)N * 256 * 2);   // bf16 features
  __bf16*   wtbf   = (__bf16*)  take((size_t)512 * 256 * 2); // bf16 W^T
  float*    es     = (float*)   take((size_t)N * 4 * 4);
  float*    ed     = (float*)   take((size_t)N * 4 * 4);
  unsigned* emax   = (unsigned*)take((size_t)N * 4 * 4);
  float*    denom  = (float*)   take((size_t)N * 4 * 4);
  float*    exf    = (float*)   take((size_t)EE * 4 * 4);
  float*    pooled = (float*)   take((size_t)G * OUT * 4);
  float*    cntb   = (float*)   take((size_t)G * 4);
  float*    z1     = (float*)   take((size_t)G * OUT * 4);
  float*    z2     = (float*)   take((size_t)G * OUT * 4);
  (void)ws_size; (void)n_in; (void)out_size;

  const int Kd[4]  = {8, 256, 256, 256};
  const int Kp[4]  = {32, 256, 256, 256};
  const int Cd[4]  = {64, 64, 64, 128};
  const int HCd[4] = {256, 256, 256, 512};

  for (int l = 0; l < 4; ++l) {
    const int K = Kd[l], KP = Kp[l], C = Cd[l], HC = HCd[l];

    if (l == 0)  // later layers: bf16 features produced by previous bias_act
      cvt_feats<<<gblocks((long long)N * KP), WG, 0, stream>>>(x, hbf, N, K, KP);
    cvt_weights<<<gblocks((long long)HC * KP), WG, 0, stream>>>(W_[l], wtbf, K, KP, HC);

    long long waves = (long long)(N / 16) * (HC / 64);
    gemm_bf16_wmma<<<gblocks(waves * 32), WG, 0, stream>>>(hbf, wtbf, hB, N, KP, HC);

    node_attn<<<gblocks((long long)N * 4), WG, 0, stream>>>(hB, As[l], Ad[l], es, ed, N, C, HC);

    fill_u32<<<gblocks((long long)N * 4), WG, 0, stream>>>(emax, 0x007FFFFFu, (long long)N * 4); // ord(-inf)
    fill_f32<<<gblocks((long long)N * 4), WG, 0, stream>>>(denom, 0.f, (long long)N * 4);
    fill_f32<<<gblocks((long long)N * HC), WG, 0, stream>>>(hA, 0.f, (long long)N * HC);

    edge_max  <<<gblocks(EE * 4), WG, 0, stream>>>(ei, E, EE, es, ed, emax);
    edge_exp  <<<gblocks(EE * 4), WG, 0, stream>>>(ei, E, EE, es, ed, emax, denom, exf);
    edge_alpha<<<gblocks(EE * 4), WG, 0, stream>>>(ei, E, EE, denom, exf);
    edge_agg  <<<gblocks(EE * 4 * (C / 4)), WG, 0, stream>>>(ei, E, EE, C, HC, exf, hB, hA);

    bias_act<<<gblocks((long long)N * HC), WG, 0, stream>>>(
        hA, Bs[l], (l < 3) ? hbf : (__bf16*)nullptr,
        (long long)N * HC, HC, l < 3 ? 1 : 0);
  }

  // mean pool per graph + LN1 + FC(relu) + LN2
  fill_f32<<<gblocks((long long)G * OUT), WG, 0, stream>>>(pooled, 0.f, (long long)G * OUT);
  fill_f32<<<gblocks(G), WG, 0, stream>>>(cntb, 0.f, G);
  pool_cnt<<<gblocks(N), WG, 0, stream>>>(batch, cntb, N);
  pool_add<<<gblocks((long long)N * 128), WG, 0, stream>>>(hA, batch, pooled, N);

  ln_row<<<G, 256, 0, stream>>>(pooled, cntb, ln1g, ln1b, z1, OUT);
  fc_relu<<<gblocks(G * OUT), WG, 0, stream>>>(z1, fcW, fcb, z2, OUT);
  ln_row<<<G, 256, 0, stream>>>(z2, nullptr, ln2g, ln2b, (float*)d_out, OUT);
}
